// BaseGenerator_38989713113442
// MI455X (gfx1250) — compile-verified, matrix operand order checked by reading, not run on hardware
//
#include <hip/hip_runtime.h>

// ---------------- types ----------------
typedef __attribute__((ext_vector_type(16))) __bf16 v16bf;
typedef __attribute__((ext_vector_type(8)))  float  v8f;
typedef __attribute__((ext_vector_type(4)))  unsigned v4u;
typedef __attribute__((ext_vector_type(8)))  unsigned v8u;

#define NEGF (-__builtin_huge_valf())

// model constants
#define Bn 8
#define Ln 512
#define EMB 512
#define NHEAD 8
#define HD 64
#define FFN 2048
#define NL 6
#define VOCAB 45
#define NRE 20
#define RING_START 24
#define PAD_TOK 0
#define NOUT0 (VOCAB - NRE)   // 25
#define Mrows (Bn * Ln)       // 4096
#define SQRT_EMB 22.627416997969522f
#define INV_SQRT_HD 0.125f
#define INV_SQRT_EMB 0.04419417382415922f

// ---------------- WMMA tile helpers ----------------
// LDS tile stride (bf16 elems): 32 payload + 8 pad (=16+4 DWORDs, matches TDM pad)
#define TS 40

// ---- Tensor Data Mover: DMA one 64-row x 32-col bf16 tile into LDS ----
// D# group0 (4 SGPRs) + group1 (8 SGPRs); 2D tile, data_size=2B,
// pad_enable with pad_interval=16 DWORDs (code 3) and pad_amount=4 DWORDs
// (code 3) -> LDS row stride of 40 halves. Rows beyond rows_avail read zero.
__device__ __forceinline__ unsigned lds_off(const void* p) {
  return (unsigned)(unsigned long long)p;  // generic shared ptr low 32 = LDS offset
}

__device__ __forceinline__ void tdm_load_tile(const __bf16* gbase,
                                              int stride_elems, int rows_avail,
                                              unsigned ldsaddr) {
  const unsigned long long ga = (unsigned long long)gbase;
  const unsigned td0 = (unsigned)stride_elems;          // tensor_dim0
  const unsigned td1 = (unsigned)(rows_avail > 0 ? rows_avail : 0);
  v4u g0;
  g0[0] = 1u;                                           // count=1, user mode
  g0[1] = ldsaddr;                                      // lds_addr (bytes)
  g0[2] = (unsigned)ga;                                 // global_addr[31:0]
  g0[3] = (unsigned)((ga >> 32) & 0x01FFFFFFu) | (2u << 30);  // [56:32] | type=2
  v8u g1;
  g1[0] = (1u << 16) | (1u << 20) | (3u << 22) | (3u << 25);  // 2B, pad 16DW+4DW
  g1[1] = (td0 & 0xFFFFu) << 16;                        // abar=0, td0[15:0]
  g1[2] = (td0 >> 16) | ((td1 & 0xFFFFu) << 16);        // td0[31:16], td1[15:0]
  g1[3] = (td1 >> 16) | (32u << 16);                    // td1[31:16], tile_dim0=32
  g1[4] = 64u;                                          // tile_dim1=64, tile_dim2=0
  g1[5] = (unsigned)stride_elems;                       // dim0_stride[31:0]
  g1[6] = 0u;                                           // stride hi, dim1_stride lo
  g1[7] = 0u;
  asm volatile("tensor_load_to_lds %0, %1" :: "s"(g0), "s"(g1) : "memory");
}

__device__ __forceinline__ void tdm_wait() {
  __builtin_amdgcn_s_wait_tensorcnt(0);
}

// Gather one 16x32 bf16 fragment from an LDS tile (stride TS), following the
// CDNA5 16-bit A/B operand VGPR layout.
__device__ __forceinline__ v16bf frag_ld(const __bf16* t, int lane) {
  const int r  = (lane & 15) * TS;
  const int kb = (lane >> 4) << 3;
  v16bf f;
#pragma unroll
  for (int j = 0; j < 8; ++j) {
    const int k = ((j & 4) << 2) + kb + ((j & 3) << 1);
    f[2 * j]     = t[r + k];
    f[2 * j + 1] = t[r + k + 1];
  }
  return f;
}

// One K-step (32) of a 64x64 block: 4 waves, each wave a 32x32 subtile
// as 2x2 v_wmma_f32_16x16x32_bf16.
__device__ __forceinline__ void wmma_step(const __bf16* As, const __bf16* Ws,
                                          int wm, int wn, int lane,
                                          v8f acc[2][2]) {
  v16bf a[2], b[2];
#pragma unroll
  for (int i = 0; i < 2; ++i) a[i] = frag_ld(&As[(wm + 16 * i) * TS], lane);
#pragma unroll
  for (int j = 0; j < 2; ++j) b[j] = frag_ld(&Ws[(wn + 16 * j) * TS], lane);
#pragma unroll
  for (int i = 0; i < 2; ++i)
#pragma unroll
    for (int j = 0; j < 2; ++j)
      acc[i][j] = __builtin_amdgcn_wmma_f32_16x16x32_bf16(
          false, a[i], false, b[j], (short)0, acc[i][j], false, false);
}

// ---------------- generic GEMM:  C[M,N] = A[M,K] * W[N,K]^T + bias ----------------
// A row-major lda, W row-major (ldw == K). M must be a multiple of 64 (true for
// every call site: M = 4096). N may be ragged (gen head N=25): TDM zero-fills.
__global__ __launch_bounds__(128) void gemm_kernel(
    const __bf16* __restrict__ A, int lda, const __bf16* __restrict__ W,
    const float* __restrict__ bias, float* __restrict__ C,
    __bf16* __restrict__ Cbf, int ldc, int M, int N, int K, int act) {
  __shared__ __align__(16) __bf16 As[64 * TS];
  __shared__ __align__(16) __bf16 Ws[64 * TS];
  const int tid = threadIdx.x, lane = tid & 31, wave = tid >> 5;
  const int wm = (wave >> 1) * 32, wn = (wave & 1) * 32;
  const int m0 = blockIdx.y * 64, n0 = blockIdx.x * 64;
  v8f acc[2][2] = {};

  for (int kt = 0; kt < K; kt += 32) {
    __syncthreads();
    if (wave == 0)
      tdm_load_tile(A + (size_t)m0 * lda + kt, lda, 64, lds_off(As));
    else if (wave == 1)
      tdm_load_tile(W + (size_t)n0 * K + kt, K, N - n0, lds_off(Ws));
    tdm_wait();
    __syncthreads();
    wmma_step(As, Ws, wm, wn, lane, acc);
  }

  const int n = lane & 15, mh = (lane >> 4) << 3;
#pragma unroll
  for (int j = 0; j < 2; ++j) {
    const int col = n0 + wn + 16 * j + n;
    if (col >= N) continue;
    const float bv = bias ? bias[col] : 0.0f;
#pragma unroll
    for (int i = 0; i < 2; ++i) {
#pragma unroll
      for (int q = 0; q < 8; ++q) {
        const int row = m0 + wm + 16 * i + mh + q;
        float v = acc[i][j][q] + bv;
        if (act == 1) v = 0.5f * v * (1.0f + erff(v * 0.7071067811865476f));
        if (C)   C[(size_t)row * ldc + col] = v;
        if (Cbf) Cbf[(size_t)row * ldc + col] = (__bf16)v;
      }
    }
  }
}

// ---------------- attention scores: S = (Q K^T)/sqrt(HD) + bias ----------------
__global__ __launch_bounds__(128) void scores_kernel(
    const __bf16* __restrict__ qkv, const float* __restrict__ attn_bias,
    float* __restrict__ scores) {
  __shared__ __align__(16) __bf16 Qs[64 * TS];
  __shared__ __align__(16) __bf16 Ks[64 * TS];
  const int tid = threadIdx.x, lane = tid & 31, wave = tid >> 5;
  const int wm = (wave >> 1) * 32, wn = (wave & 1) * 32;
  const int z = blockIdx.z;           // b*NHEAD + h
  const int b = z >> 3, h = z & 7;
  const int q0 = blockIdx.y * 64, k0 = blockIdx.x * 64;
  const __bf16* Qb = qkv + (size_t)b * Ln * (3 * EMB) + h * HD;
  const __bf16* Kb = Qb + EMB;
  v8f acc[2][2] = {};

  for (int kt = 0; kt < HD; kt += 32) {
    __syncthreads();
    if (wave == 0)
      tdm_load_tile(Qb + (size_t)q0 * (3 * EMB) + kt, 3 * EMB, 64, lds_off(Qs));
    else if (wave == 1)
      tdm_load_tile(Kb + (size_t)k0 * (3 * EMB) + kt, 3 * EMB, 64, lds_off(Ks));
    tdm_wait();
    __syncthreads();
    wmma_step(Qs, Ks, wm, wn, lane, acc);
  }

  const int n = lane & 15, mh = (lane >> 4) << 3;
  const size_t zb = (size_t)z * Ln * Ln;
#pragma unroll
  for (int i = 0; i < 2; ++i)
#pragma unroll
    for (int j = 0; j < 2; ++j)
#pragma unroll
      for (int q = 0; q < 8; ++q) {
        const int row = q0 + wm + 16 * i + mh + q;
        const int col = k0 + wn + 16 * j + n;
        const size_t idx = zb + (size_t)row * Ln + col;
        scores[idx] = acc[i][j][q] * INV_SQRT_HD + attn_bias[idx];
      }
}

// ---------------- attention context: ctx = P @ V  (per b,h) ----------------
__global__ __launch_bounds__(128) void ctx_kernel(
    const __bf16* __restrict__ probs, const __bf16* __restrict__ qkv,
    __bf16* __restrict__ ctxbf) {
  __shared__ __align__(16) __bf16 Ps[64 * TS];
  __shared__ __align__(16) __bf16 Vs[64 * TS];  // [d][k] transposed stage
  const int tid = threadIdx.x, lane = tid & 31, wave = tid >> 5;
  const int wm = (wave >> 1) * 32, wn = (wave & 1) * 32;
  const int z = blockIdx.y;  // b*NHEAD + h
  const int b = z >> 3, h = z & 7;
  const int q0 = blockIdx.x * 64;
  const __bf16* Pb = probs + (size_t)z * Ln * Ln;
  const __bf16* Vb = qkv + (size_t)b * Ln * (3 * EMB) + 2 * EMB + h * HD;
  v8f acc[2][2] = {};

  for (int kt = 0; kt < Ln; kt += 32) {
    __syncthreads();
    if (wave == 0)
      tdm_load_tile(Pb + (size_t)q0 * Ln + kt, Ln, 64, lds_off(Ps));
    {  // V staged transposed: Vs[d][k] = V[kt+k][d]
      const int d = tid >> 1, kc = (tid & 1) * 16;
#pragma unroll
      for (int kk = 0; kk < 16; ++kk)
        Vs[d * TS + kc + kk] = Vb[(size_t)(kt + kc + kk) * (3 * EMB) + d];
    }
    tdm_wait();
    __syncthreads();
    wmma_step(Ps, Vs, wm, wn, lane, acc);
  }

  const int n = lane & 15, mh = (lane >> 4) << 3;
#pragma unroll
  for (int i = 0; i < 2; ++i)
#pragma unroll
    for (int j = 0; j < 2; ++j)
#pragma unroll
      for (int q = 0; q < 8; ++q) {
        const int row = q0 + wm + 16 * i + mh + q;  // query index
        const int col = wn + 16 * j + n;            // d within head
        ctxbf[((size_t)(b * Ln + row)) * EMB + h * HD + col] =
            (__bf16)acc[i][j][q];
      }
}

// ---------------- elementwise / reduction kernels ----------------
__global__ void convert_bf16_kernel(const float* __restrict__ in,
                                    __bf16* __restrict__ out, long n) {
  long i = (long)blockIdx.x * blockDim.x + threadIdx.x;
  if (i < n) out[i] = (__bf16)in[i];
}

__global__ void zero_kernel(float* __restrict__ p, long n) {
  long i = (long)blockIdx.x * blockDim.x + threadIdx.x;
  if (i < n) p[i] = 0.0f;
}

__global__ void embed_kernel(const int* __restrict__ seq,
                             const int* __restrict__ cnt,
                             const float* __restrict__ tok,
                             const float* __restrict__ cnte,
                             float* __restrict__ x, __bf16* __restrict__ xbf) {
  long i = (long)blockIdx.x * blockDim.x + threadIdx.x;
  if (i >= (long)Mrows * EMB) return;
  const int e = (int)(i % EMB);
  const long m = i / EMB;
  const float v =
      (tok[(size_t)seq[m] * EMB + e] + cnte[(size_t)cnt[m] * EMB + e]) *
      SQRT_EMB;
  x[i] = v;
  xbf[i] = (__bf16)v;
}

__global__ void bias_kernel(const int* __restrict__ lin_sq,
                            const int* __restrict__ up_sq,
                            const int* __restrict__ dn_sq,
                            const int* __restrict__ seq,
                            const float* __restrict__ lin_e,
                            const float* __restrict__ up_e,
                            const float* __restrict__ dn_e,
                            float* __restrict__ bias) {
  long i = (long)blockIdx.x * blockDim.x + threadIdx.x;
  if (i >= (long)Bn * Ln * Ln) return;
  const int b = (int)(i / (Ln * Ln));
  const int r = (int)(i % (Ln * Ln));
  const int q = r / Ln, k = r % Ln;
  const bool masked = (k > q) || (seq[b * Ln + k] == PAD_TOK);
  const int a = lin_sq[i], u = up_sq[i], d = dn_sq[i];
#pragma unroll
  for (int h = 0; h < NHEAD; ++h) {
    const size_t o = (((size_t)(b * NHEAD + h)) * Ln + q) * Ln + k;
    bias[o] = masked ? NEGF
                     : lin_e[a * NHEAD + h] + up_e[u * NHEAD + h] +
                           dn_e[d * NHEAD + h];
  }
}

// wave-per-row softmax over 512, bf16 probabilities out
__global__ __launch_bounds__(256) void softmax_kernel(
    const float* __restrict__ scores, __bf16* __restrict__ probs) {
  const int row = blockIdx.x * 8 + (threadIdx.x >> 5);
  const int lane = threadIdx.x & 31;
  const float* s = scores + (size_t)row * Ln;
  float vals[16], mx = NEGF;
#pragma unroll
  for (int t = 0; t < 16; ++t) {
    vals[t] = s[lane + t * 32];
    mx = fmaxf(mx, vals[t]);
  }
#pragma unroll
  for (int o = 16; o > 0; o >>= 1) mx = fmaxf(mx, __shfl_xor(mx, o, 32));
  float sum = 0.0f;
  const bool dead = (mx == NEGF);
#pragma unroll
  for (int t = 0; t < 16; ++t) {
    vals[t] = dead ? 0.0f : __expf(vals[t] - mx);
    sum += vals[t];
  }
#pragma unroll
  for (int o = 16; o > 0; o >>= 1) sum += __shfl_xor(sum, o, 32);
  const float inv = (sum > 0.0f) ? (1.0f / sum) : 0.0f;
#pragma unroll
  for (int t = 0; t < 16; ++t)
    probs[(size_t)row * Ln + lane + t * 32] = (__bf16)(vals[t] * inv);
}

// wave-per-row LayerNorm of (xin + delta), writes fp32 (opt) and bf16 (opt)
__global__ __launch_bounds__(256) void ln_kernel(
    const float* __restrict__ xin, const float* __restrict__ delta,
    const float* __restrict__ g, const float* __restrict__ bta,
    float* __restrict__ xout, __bf16* __restrict__ xbf) {
  const int row = blockIdx.x * 8 + (threadIdx.x >> 5);
  const int lane = threadIdx.x & 31;
  const size_t base = (size_t)row * EMB;
  float vals[16], sum = 0.0f, sq = 0.0f;
#pragma unroll
  for (int t = 0; t < 16; ++t) {
    const int e = lane + t * 32;
    float v = xin[base + e];
    if (delta) v += delta[base + e];
    vals[t] = v;
    sum += v;
    sq += v * v;
  }
#pragma unroll
  for (int o = 16; o > 0; o >>= 1) {
    sum += __shfl_xor(sum, o, 32);
    sq += __shfl_xor(sq, o, 32);
  }
  const float mean = sum * (1.0f / EMB);
  const float var = sq * (1.0f / EMB) - mean * mean;
  const float rs = rsqrtf(var + 1e-5f);
#pragma unroll
  for (int t = 0; t < 16; ++t) {
    const int e = lane + t * 32;
    const float o = (vals[t] - mean) * rs * g[e] + bta[e];
    if (xout) xout[base + e] = o;
    if (xbf) xbf[base + e] = (__bf16)o;
  }
}

// ring cumsum scatter: one block per batch
__global__ __launch_bounds__(64) void ring_scan_kernel(
    const int* __restrict__ seq, const float* __restrict__ out1,
    float* __restrict__ ring_tab) {
  __shared__ int sidx[Ln];
  const int b = blockIdx.x;
  if (threadIdx.x == 0) {
    int c = 0;
    for (int l = 0; l < Ln; ++l) {
      int v = -1;
      if (seq[b * Ln + l] == RING_START) {
        ++c;
        if (c <= NRE) v = c - 1;
      }
      sidx[l] = v;
    }
  }
  __syncthreads();
  for (int l = 0; l < Ln; ++l) {
    const int s = sidx[l];
    if (s < 0) continue;
    for (int e = threadIdx.x; e < EMB; e += 64)
      ring_tab[((size_t)b * NRE + s) * EMB + e] =
          out1[((size_t)b * Ln + l) * EMB + e];
  }
}

__global__ void logits1_kernel(const float* __restrict__ out0,
                               const float* __restrict__ ring_tab,
                               float* __restrict__ logits1) {
  long i = (long)blockIdx.x * blockDim.x + threadIdx.x;
  if (i >= (long)Bn * Ln * NRE) return;
  const int b = (int)(i / (Ln * NRE));
  const int r2 = (int)(i % (Ln * NRE));
  const int l = r2 / NRE, r = r2 % NRE;
  const float4* a = (const float4*)(out0 + ((size_t)b * Ln + l) * EMB);
  const float4* t = (const float4*)(ring_tab + ((size_t)b * NRE + r) * EMB);
  float s = 0.0f;
#pragma unroll 4
  for (int j = 0; j < EMB / 4; ++j) {
    const float4 av = a[j], tv = t[j];
    s += av.x * tv.x + av.y * tv.y + av.z * tv.z + av.w * tv.w;
  }
  logits1[i] = s * INV_SQRT_EMB;
}

__global__ void final_kernel(const float* __restrict__ logits0,
                             const float* __restrict__ logits1,
                             const unsigned char* __restrict__ gmask,
                             const unsigned char* __restrict__ vmask,
                             float* __restrict__ out) {
  long i = (long)blockIdx.x * blockDim.x + threadIdx.x;
  if (i >= (long)Mrows * VOCAB) return;
  const int o = (int)(i % VOCAB);
  const long ml = i / VOCAB;
  float v = (o < NOUT0) ? logits0[ml * NOUT0 + o] : logits1[ml * NRE + (o - NOUT0)];
  if (gmask[i] || vmask[i]) v = NEGF;
  out[i] = v;
}

// ---------------- host launcher ----------------
static inline dim3 gemm_grid(int M, int N) {
  return dim3((N + 63) / 64, (M + 63) / 64, 1);
}

extern "C" void kernel_launch(void* const* d_in, const int* in_sizes, int n_in,
                              void* d_out, int out_size, void* d_ws,
                              size_t ws_size, hipStream_t stream) {
  // inputs (setup_inputs dict order)
  const int* seq = (const int*)d_in[0];
  const int* cnt = (const int*)d_in[1];
  const unsigned char* gmask = (const unsigned char*)d_in[2];
  const unsigned char* vmask = (const unsigned char*)d_in[3];
  const int* lin_sq = (const int*)d_in[4];
  const int* up_sq = (const int*)d_in[5];
  const int* dn_sq = (const int*)d_in[6];
  const float* tok_emb = (const float*)d_in[7];
  const float* cnt_emb = (const float*)d_in[8];
  const float* lin_e = (const float*)d_in[9];
  const float* up_e = (const float*)d_in[10];
  const float* dn_e = (const float*)d_in[11];
  const float* in_proj_w = (const float*)d_in[12];
  const float* in_proj_b = (const float*)d_in[13];
  const float* out_w = (const float*)d_in[14];
  const float* out_b = (const float*)d_in[15];
  const float* ln1_g = (const float*)d_in[16];
  const float* ln1_b = (const float*)d_in[17];
  const float* ln2_g = (const float*)d_in[18];
  const float* ln2_b = (const float*)d_in[19];
  const float* ffn_w1 = (const float*)d_in[20];
  const float* ffn_b1 = (const float*)d_in[21];
  const float* ffn_w2 = (const float*)d_in[22];
  const float* ffn_b2 = (const float*)d_in[23];
  const float* fin_g = (const float*)d_in[24];
  const float* fin_b = (const float*)d_in[25];
  const float* gen_w = (const float*)d_in[26];
  const float* gen_b = (const float*)d_in[27];
  const float* ring_w0 = (const float*)d_in[28];
  const float* ring_b0 = (const float*)d_in[29];
  const float* ring_w1 = (const float*)d_in[30];
  const float* ring_b1 = (const float*)d_in[31];

  // workspace carving
  size_t cur = 0;
  auto alloc = [&](size_t bytes) -> void* {
    cur = (cur + 255) & ~(size_t)255;
    void* p = (char*)d_ws + cur;
    cur += bytes;
    return p;
  };
  const long n_inproj = (long)NL * 3 * EMB * EMB;
  const long n_outw = (long)NL * EMB * EMB;
  const long n_f1 = (long)NL * FFN * EMB;
  const long n_f2 = (long)NL * EMB * FFN;
  __bf16* w_inproj = (__bf16*)alloc(n_inproj * 2);
  __bf16* w_out = (__bf16*)alloc(n_outw * 2);
  __bf16* w_f1 = (__bf16*)alloc(n_f1 * 2);
  __bf16* w_f2 = (__bf16*)alloc(n_f2 * 2);
  __bf16* w_gen = (__bf16*)alloc((long)NOUT0 * EMB * 2);
  __bf16* w_r0 = (__bf16*)alloc((long)EMB * EMB * 2);
  __bf16* w_r1 = (__bf16*)alloc((long)EMB * EMB * 2);
  float* x = (float*)alloc((size_t)Mrows * EMB * 4);
  __bf16* xbf = (__bf16*)alloc((size_t)Mrows * EMB * 2);
  __bf16* qkvbf = (__bf16*)alloc((size_t)Mrows * 3 * EMB * 2);
  float* attn_bias = (float*)alloc((size_t)Bn * NHEAD * Ln * Ln * 4);
  float* scores = (float*)alloc((size_t)Bn * NHEAD * Ln * Ln * 4);
  __bf16* probsbf = (__bf16*)alloc((size_t)Bn * NHEAD * Ln * Ln * 2);
  __bf16* ctxbf = (__bf16*)alloc((size_t)Mrows * EMB * 2);
  float* attn_out = (float*)alloc((size_t)Mrows * EMB * 4);
  __bf16* hbf = (__bf16*)alloc((size_t)Mrows * FFN * 2);
  float* ff = (float*)alloc((size_t)Mrows * EMB * 4);
  __bf16* xfbf = (__bf16*)alloc((size_t)Mrows * EMB * 2);
  float* logits0 = (float*)alloc((size_t)Mrows * NOUT0 * 4);
  float* out0 = (float*)alloc((size_t)Mrows * EMB * 4);
  float* out1 = (float*)alloc((size_t)Mrows * EMB * 4);
  float* ring_tab = (float*)alloc((size_t)Bn * NRE * EMB * 4);
  float* logits1 = (float*)alloc((size_t)Mrows * NRE * 4);

  const int TB = 256;
  auto nb = [&](long n) { return (unsigned)((n + TB - 1) / TB); };

  // weight conversion fp32 -> bf16 (every call; deterministic)
  convert_bf16_kernel<<<nb(n_inproj), TB, 0, stream>>>(in_proj_w, w_inproj, n_inproj);
  convert_bf16_kernel<<<nb(n_outw), TB, 0, stream>>>(out_w, w_out, n_outw);
  convert_bf16_kernel<<<nb(n_f1), TB, 0, stream>>>(ffn_w1, w_f1, n_f1);
  convert_bf16_kernel<<<nb(n_f2), TB, 0, stream>>>(ffn_w2, w_f2, n_f2);
  convert_bf16_kernel<<<nb((long)NOUT0 * EMB), TB, 0, stream>>>(gen_w, w_gen, (long)NOUT0 * EMB);
  convert_bf16_kernel<<<nb((long)EMB * EMB), TB, 0, stream>>>(ring_w0, w_r0, (long)EMB * EMB);
  convert_bf16_kernel<<<nb((long)EMB * EMB), TB, 0, stream>>>(ring_w1, w_r1, (long)EMB * EMB);

  // embedding + attention bias (bias reused by all layers)
  embed_kernel<<<nb((long)Mrows * EMB), TB, 0, stream>>>(seq, cnt, tok_emb, cnt_emb, x, xbf);
  bias_kernel<<<nb((long)Bn * Ln * Ln), TB, 0, stream>>>(lin_sq, up_sq, dn_sq, seq, lin_e, up_e, dn_e, attn_bias);

  const dim3 blk128(128, 1, 1);
  for (int i = 0; i < NL; ++i) {
    // QKV projection (bf16 out only)
    gemm_kernel<<<gemm_grid(Mrows, 3 * EMB), blk128, 0, stream>>>(
        xbf, EMB, w_inproj + (size_t)i * 3 * EMB * EMB, in_proj_b + (size_t)i * 3 * EMB,
        nullptr, qkvbf, 3 * EMB, Mrows, 3 * EMB, EMB, 0);
    // scores = QK^T/sqrt(hd) + bias
    scores_kernel<<<dim3(Ln / 64, Ln / 64, Bn * NHEAD), blk128, 0, stream>>>(
        qkvbf, attn_bias, scores);
    // softmax -> bf16 probs
    softmax_kernel<<<(Bn * NHEAD * Ln) / 8, 256, 0, stream>>>(scores, probsbf);
    // ctx = P @ V
    ctx_kernel<<<dim3(Ln / 64, Bn * NHEAD, 1), blk128, 0, stream>>>(probsbf, qkvbf, ctxbf);
    // out projection (fp32 out)
    gemm_kernel<<<gemm_grid(Mrows, EMB), blk128, 0, stream>>>(
        ctxbf, EMB, w_out + (size_t)i * EMB * EMB, out_b + (size_t)i * EMB,
        attn_out, nullptr, EMB, Mrows, EMB, EMB, 0);
    // x = LN(x + attn_out)
    ln_kernel<<<Mrows / 8, 256, 0, stream>>>(x, attn_out, ln1_g + (size_t)i * EMB,
                                             ln1_b + (size_t)i * EMB, x, xbf);
    // FFN up + GELU (bf16 out only)
    gemm_kernel<<<gemm_grid(Mrows, FFN), blk128, 0, stream>>>(
        xbf, EMB, w_f1 + (size_t)i * FFN * EMB, ffn_b1 + (size_t)i * FFN,
        nullptr, hbf, FFN, Mrows, FFN, EMB, 1);
    // FFN down (fp32 out)
    gemm_kernel<<<gemm_grid(Mrows, EMB), blk128, 0, stream>>>(
        hbf, FFN, w_f2 + (size_t)i * EMB * FFN, ffn_b2 + (size_t)i * EMB,
        ff, nullptr, EMB, Mrows, EMB, FFN, 0);
    // x = LN(x + ff)
    ln_kernel<<<Mrows / 8, 256, 0, stream>>>(x, ff, ln2_g + (size_t)i * EMB,
                                             ln2_b + (size_t)i * EMB, x, xbf);
  }

  // final LN (bf16 out only)
  ln_kernel<<<Mrows / 8, 256, 0, stream>>>(x, nullptr, fin_g, fin_b, nullptr, xfbf);

  // heads
  gemm_kernel<<<gemm_grid(Mrows, NOUT0), blk128, 0, stream>>>(
      xfbf, EMB, w_gen, gen_b, logits0, nullptr, NOUT0, Mrows, NOUT0, EMB, 0);
  gemm_kernel<<<gemm_grid(Mrows, EMB), blk128, 0, stream>>>(
      xfbf, EMB, w_r0, ring_b0, out0, nullptr, EMB, Mrows, EMB, EMB, 0);
  gemm_kernel<<<gemm_grid(Mrows, EMB), blk128, 0, stream>>>(
      xfbf, EMB, w_r1, ring_b1, out1, nullptr, EMB, Mrows, EMB, EMB, 0);

  // ring table + logits1
  zero_kernel<<<nb((long)Bn * NRE * EMB), TB, 0, stream>>>(ring_tab, (long)Bn * NRE * EMB);
  ring_scan_kernel<<<Bn, 64, 0, stream>>>(seq, out1, ring_tab);
  logits1_kernel<<<nb((long)Bn * Ln * NRE), TB, 0, stream>>>(out0, ring_tab, logits1);

  // concat + masks
  final_kernel<<<nb((long)Mrows * VOCAB), TB, 0, stream>>>(
      logits0, logits1, gmask, vmask, (float*)d_out);
  (void)in_sizes; (void)n_in; (void)out_size; (void)ws_size;
}
